// FrequencySplit_87754771792328
// MI455X (gfx1250) — compile-verified, matrix operand order checked by reading, not run on hardware
//
#include <hip/hip_runtime.h>
#include <hip/hip_bf16.h>

// ---------------------------------------------------------------------------
// FrequencySplit on MI455X (gfx1250, wave32, WMMA + TDM).
// conv0/conv1 are implicit GEMMs on v_wmma_f32_16x16x32_f16 with f32 accum.
// A tile = 16 pixels along W; K = tap-major (9 taps) x channel-minor (64ch).
// B is pre-packed into the exact per-lane VGPR layout (ISA 7.12.2), then
// DMA'd once per workgroup into LDS via TENSOR_LOAD_TO_LDS (TENSORcnt wait),
// and consumed with ds_load_b128. All 8 waves of a block share one B copy.
// ---------------------------------------------------------------------------

typedef __attribute__((ext_vector_type(16))) _Float16     v16h;
typedef __attribute__((ext_vector_type(8)))  _Float16     v8h;
typedef __attribute__((ext_vector_type(8)))  float        v8f;
typedef __attribute__((ext_vector_type(4)))  unsigned int v4u;
typedef __attribute__((ext_vector_type(8)))  int          v8i_;
typedef __attribute__((ext_vector_type(4)))  int          v4i_;

#define NN 8
#define CC 64
#define HH 128
#define WW 128
#define KTAPS 9
#define KCH 576        // 9*64
#define NCHUNKS 18     // KCH / 32
#define NPIX (NN*HH*WW)          // 131072
#define NELEM (NN*CC*HH*WW)      // 8388608

#define WP0_HALVES (4 * NCHUNKS * 32 * 16)   // 36864 halves = 73728 B
#define WP1_HALVES (NCHUNKS * 32 * 16)       //  9216 halves = 18432 B

// workspace layout (byte offsets, all 32B-aligned)
#define OFF_X16   0ull                     // 16,777,216 B : x NHWC f16
#define OFF_Y0    16777216ull              // 16,777,216 B : conv0 out NHWC f16
#define OFF_KERN  33554432ull              //  4,718,592 B : 9 logits/weights per pixel f32
#define OFF_WP0   38273024ull              //     73,728 B : packed w0 (4 ntiles)
#define OFF_WP1   38346752ull              //     18,432 B : packed w1 (1 ntile, cols 9..15 = 0)

#if __has_builtin(__builtin_amdgcn_tensor_load_to_lds) && __has_builtin(__builtin_amdgcn_s_wait_tensorcnt)
#define USE_TDM 1
#else
#define USE_TDM 0
#endif

// The TDM builtin's LDS destination is buried in an opaque descriptor, so LLVM
// cannot see that sB gets written and would fold the ds_loads to undef (it did:
// round-3 histogram showed ds=0 and CSE'd WMMAs). Escaping the pointer into an
// asm with a memory clobber forces the compiler to treat the buffer as written.
__device__ __forceinline__ void lds_publish(void* p) {
    asm volatile("" :: "r"(p) : "memory");
}

// ------------------------ TDM 1-D descriptor load ---------------------------
// Copies `ndwords` DWORDs from global `gsrc` into LDS offset `lds_off` with one
// Tensor Data Mover op. D# per CDNA5 ISA:
//  group0: [1:0]=count=1, [63:32]=lds_addr, [120:64]=global_addr, [127:126]=2
//  group1: data_size=4B, tensor_dim0=tile_dim0=ndwords (1 row), dim1=1
// This toolchain's builtin takes 6 args (g0 v4u, g1 v8i, g2 v4i, g3 v4i,
// extra v8i, cpol i32) -- groups 2/3 unused for a <=2D tensor, pass zeros.
#if USE_TDM
__device__ __forceinline__ void tdm_load_1d(const void* gsrc, unsigned lds_off,
                                            unsigned ndwords) {
    unsigned long long ga = (unsigned long long)gsrc;
    v4u g0 = { 1u,                                            // count=1 (user)
               lds_off,                                       // lds byte address
               (unsigned)(ga & 0xffffffffu),                  // global_addr[31:0]
               (unsigned)((ga >> 32) & 0x01ffffffu) | (2u << 30) }; // addr[56:32] | type=2
    v8i_ g1 = { (int)0x00020000u,                             // data_size=2 (4B), mask=0
                (int)((ndwords & 0xffffu) << 16),             // tensor_dim0[15:0]
                (int)(((ndwords >> 16) & 0xffffu) | (1u << 16)), // dim0[31:16] | tensor_dim1=1
                (int)((ndwords & 0xffffu) << 16),             // tile_dim0 (fits 16b here)
                1,                                            // tile_dim1=1, tile_dim2=0
                (int)ndwords,                                 // tensor_dim0_stride[31:0]
                0, 0 };
    v4i_ gz4 = { 0, 0, 0, 0 };
    v8i_ gz8 = { 0, 0, 0, 0, 0, 0, 0, 0 };
    __builtin_amdgcn_tensor_load_to_lds(g0, g1, gz4, gz4, gz8, 0);
}
#endif

// --------------------------- staging / packing ------------------------------

// x: NCHW f32 -> NHWC f16 (channels contiguous per pixel => WMMA K loads coalesce)
__global__ void k_x_to_nhwc_f16(const float* __restrict__ x, _Float16* __restrict__ x16) {
    int t = blockIdx.x * 256 + threadIdx.x;            // NCHW linear, exact grid
    int w = t & 127, h = (t >> 7) & 127, c = (t >> 14) & 63, n = t >> 20;
    x16[(((size_t)(n * HH + h) * WW + w) << 6) + c] = (_Float16)x[t];
}

// Pack w0 (OIHW 64x64x3x3) into B-operand layout:
// wp[nt][chunk][lane][j] ; lane: col N = lane%16, half = lane/16
// j<8  -> K = half*8 + j ; j>=8 -> K = 16 + half*8 + (j-8)   (K local to 32-chunk)
// chunk = tap*2 + cbase/32 ; global k = tap*64 + (cbase + Klocal) => channel index
__global__ void k_pack_w0(const float* __restrict__ w0, _Float16* __restrict__ wp) {
    int t = blockIdx.x * 256 + threadIdx.x;            // 4*18*32 = 2304 exact
    if (t >= 4 * NCHUNKS * 32) return;
    int lane = t & 31, chunk = (t >> 5) % NCHUNKS, nt = (t >> 5) / NCHUNKS;
    int half = lane >> 4, lm = lane & 15;
    int tap = chunk >> 1, cbase = (chunk & 1) * 32;
    int o = nt * 16 + lm;                              // output channel (0..63)
    v16h vals = {};
#pragma unroll
    for (int j = 0; j < 16; ++j) {
        int kl = (j < 8) ? (half * 8 + j) : (16 + half * 8 + (j - 8));
        int ci = cbase + kl;                           // input channel
        vals[j] = (_Float16)w0[(size_t)o * KCH + (size_t)ci * KTAPS + tap];
    }
    *(v16h*)(wp + (size_t)t * 16) = vals;
}

// Pack w1 (9x64x3x3), pad output channels 9..15 with zeros.
__global__ void k_pack_w1(const float* __restrict__ w1, _Float16* __restrict__ wp) {
    int t = blockIdx.x * 256 + threadIdx.x;            // need 18*32 = 576
    if (t >= NCHUNKS * 32) return;
    int lane = t & 31, chunk = t >> 5;
    int half = lane >> 4, lm = lane & 15;
    int tap = chunk >> 1, cbase = (chunk & 1) * 32;
    v16h vals = {};
    if (lm < KTAPS) {
#pragma unroll
        for (int j = 0; j < 16; ++j) {
            int kl = (j < 8) ? (half * 8 + j) : (16 + half * 8 + (j - 8));
            int ci = cbase + kl;
            vals[j] = (_Float16)w1[(size_t)lm * KCH + (size_t)ci * KTAPS + tap];
        }
    }
    *(v16h*)(wp + (size_t)t * 16) = vals;
}

// ------------------------------- conv0 --------------------------------------
// One wave = one 16-pixel M-tile x all 64 output channels (4 N-tiles).
// Block = 8 waves = one image row (W=128). Grid = N*H = 1024 rows.
// B pack (73,728 B) staged once per block into LDS (320KB/WGP on CDNA5).
__global__ __launch_bounds__(256) void k_conv0(const _Float16* __restrict__ x16,
                                               const _Float16* __restrict__ wp0,
                                               const float* __restrict__ b0,
                                               _Float16* __restrict__ y0) {
    __shared__ _Float16 sB[WP0_HALVES];    // only shared object => LDS offset 0

#if USE_TDM
    if (threadIdx.x < 32) {                // wave 0 issues the DMA (EXEC-independent)
        tdm_load_1d(wp0, 0u, WP0_HALVES / 2);
        __builtin_amdgcn_s_wait_tensorcnt(0);
    }
    __syncthreads();
#else
    for (int k = threadIdx.x; k < WP0_HALVES / 8; k += 256)   // 16B per iter
        ((v4u*)sB)[k] = ((const v4u*)wp0)[k];
    __syncthreads();
#endif
    lds_publish(sB);                       // sB now visibly "written" to the compiler

    int row = blockIdx.x;                  // n*H + h
    int n = row >> 7, h = row & 127;
    int lane = threadIdx.x & 31;
    int half = lane >> 4, lm = lane & 15;
    int wbase = (threadIdx.x >> 5) * 16;   // wave -> 16-pixel tile along W

    v8f acc0 = {}, acc1 = {}, acc2 = {}, acc3 = {};

    for (int tap = 0; tap < KTAPS; ++tap) {
        int dy = tap / 3 - 1, dx = tap % 3 - 1;
        int hy = h + dy;
        int wx = wbase + lm + dx;          // this lane's pixel (row M = lm)
        bool valid = (hy >= 0) && (hy < HH) && (wx >= 0) && (wx < WW);
        const _Float16* pa = x16 + (((size_t)(n * HH + hy) * WW + wx) << 6);
#pragma unroll
        for (int cb = 0; cb < 64; cb += 32) {
            v16h a = {};                   // zero-fill = SAME padding
            if (valid) {
                v8h lo = *(const v8h*)(pa + cb + half * 8);
                v8h hi = *(const v8h*)(pa + cb + 16 + half * 8);
#pragma unroll
                for (int j = 0; j < 8; ++j) { a[j] = lo[j]; a[8 + j] = hi[j]; }
            }
            int chunk = tap * 2 + (cb >> 5);
            const _Float16* pb = sB + ((size_t)chunk * 32 + lane) * 16;
            v16h bv0 = *(const v16h*)(pb);
            v16h bv1 = *(const v16h*)(pb + (size_t)1 * NCHUNKS * 32 * 16);
            v16h bv2 = *(const v16h*)(pb + (size_t)2 * NCHUNKS * 32 * 16);
            v16h bv3 = *(const v16h*)(pb + (size_t)3 * NCHUNKS * 32 * 16);
            acc0 = __builtin_amdgcn_wmma_f32_16x16x32_f16(false, a, false, bv0, (short)0, acc0, false, false);
            acc1 = __builtin_amdgcn_wmma_f32_16x16x32_f16(false, a, false, bv1, (short)0, acc1, false, false);
            acc2 = __builtin_amdgcn_wmma_f32_16x16x32_f16(false, a, false, bv2, (short)0, acc2, false, false);
            acc3 = __builtin_amdgcn_wmma_f32_16x16x32_f16(false, a, false, bv3, (short)0, acc3, false, false);
        }
    }

    // epilogue: bias + ReLU + store f16 NHWC. C layout: VGPR r -> M = r + 8*half, N = lm.
    size_t rowbase = (size_t)(n * HH + h) * WW;
#pragma unroll
    for (int r = 0; r < 8; ++r) {
        int wpix = wbase + r + half * 8;
        size_t pbase = (rowbase + wpix) << 6;
        float v0 = acc0[r] + b0[lm];        v0 = v0 > 0.f ? v0 : 0.f;
        float v1 = acc1[r] + b0[16 + lm];   v1 = v1 > 0.f ? v1 : 0.f;
        float v2 = acc2[r] + b0[32 + lm];   v2 = v2 > 0.f ? v2 : 0.f;
        float v3 = acc3[r] + b0[48 + lm];   v3 = v3 > 0.f ? v3 : 0.f;
        y0[pbase + lm]      = (_Float16)v0;
        y0[pbase + 16 + lm] = (_Float16)v1;
        y0[pbase + 32 + lm] = (_Float16)v2;
        y0[pbase + 48 + lm] = (_Float16)v3;
    }
}

// ------------------------------- conv1 --------------------------------------
// Same GEMM structure, single 16-col N-tile (cols 0..8 valid). Writes logits f32.
__global__ __launch_bounds__(256) void k_conv1(const _Float16* __restrict__ y0,
                                               const _Float16* __restrict__ wp1,
                                               const float* __restrict__ b1,
                                               float* __restrict__ kern) {
    __shared__ _Float16 sB[WP1_HALVES];    // only shared object => LDS offset 0

#if USE_TDM
    if (threadIdx.x < 32) {
        tdm_load_1d(wp1, 0u, WP1_HALVES / 2);
        __builtin_amdgcn_s_wait_tensorcnt(0);
    }
    __syncthreads();
#else
    for (int k = threadIdx.x; k < WP1_HALVES / 8; k += 256)
        ((v4u*)sB)[k] = ((const v4u*)wp1)[k];
    __syncthreads();
#endif
    lds_publish(sB);

    int row = blockIdx.x;
    int n = row >> 7, h = row & 127;
    int lane = threadIdx.x & 31;
    int half = lane >> 4, lm = lane & 15;
    int wbase = (threadIdx.x >> 5) * 16;

    v8f acc = {};

    for (int tap = 0; tap < KTAPS; ++tap) {
        int dy = tap / 3 - 1, dx = tap % 3 - 1;
        int hy = h + dy;
        int wx = wbase + lm + dx;
        bool valid = (hy >= 0) && (hy < HH) && (wx >= 0) && (wx < WW);
        const _Float16* pa = y0 + (((size_t)(n * HH + hy) * WW + wx) << 6);
#pragma unroll
        for (int cb = 0; cb < 64; cb += 32) {
            v16h a = {};
            if (valid) {
                v8h lo = *(const v8h*)(pa + cb + half * 8);
                v8h hi = *(const v8h*)(pa + cb + 16 + half * 8);
#pragma unroll
                for (int j = 0; j < 8; ++j) { a[j] = lo[j]; a[8 + j] = hi[j]; }
            }
            int chunk = tap * 2 + (cb >> 5);
            v16h bv = *(const v16h*)(sB + ((size_t)chunk * 32 + lane) * 16);
            acc = __builtin_amdgcn_wmma_f32_16x16x32_f16(false, a, false, bv, (short)0, acc, false, false);
        }
    }

    size_t rowbase = (size_t)(n * HH + h) * WW;
    if (lm < KTAPS) {
        float bb = b1[lm];
#pragma unroll
        for (int r = 0; r < 8; ++r) {
            int wpix = wbase + r + half * 8;
            kern[(rowbase + wpix) * KTAPS + lm] = acc[r] + bb;
        }
    }
}

// ------------------------- softmax over the 9 taps --------------------------
__global__ void k_softmax9(float* __restrict__ kern) {
    int pix = blockIdx.x * 256 + threadIdx.x;          // 131072 exact
    float v[KTAPS];
    float m = -3.0e38f;
#pragma unroll
    for (int k = 0; k < KTAPS; ++k) { v[k] = kern[(size_t)pix * KTAPS + k]; m = v[k] > m ? v[k] : m; }
    float s = 0.f;
#pragma unroll
    for (int k = 0; k < KTAPS; ++k) { v[k] = __expf(v[k] - m); s += v[k]; }
    float inv = 1.f / s;
#pragma unroll
    for (int k = 0; k < KTAPS; ++k) kern[(size_t)pix * KTAPS + k] = v[k] * inv;
}

// ---------------- dynamic per-pixel low-pass filter + split -----------------
__global__ void k_filter(const float* __restrict__ x, const float* __restrict__ kern,
                         float* __restrict__ out) {
    int t = blockIdx.x * 256 + threadIdx.x;            // NCHW linear, 8388608 exact
    int w = t & 127, h = (t >> 7) & 127, c = (t >> 14) & 63, n = t >> 20;
    const float* kw = kern + (size_t)((n * HH + h) * WW + w) * KTAPS;
    float l = 0.f;
#pragma unroll
    for (int i = 0; i < 3; ++i) {
        int hy = h + i - 1;
        if (hy < 0 || hy >= HH) continue;
        size_t base = ((size_t)(n * CC + c) * HH + hy) * WW;
#pragma unroll
        for (int j = 0; j < 3; ++j) {
            int wx = w + j - 1;
            if (wx < 0 || wx >= WW) continue;
            l += x[base + wx] * kw[i * 3 + j];
        }
    }
    out[t] = l;                                        // low-pass
    out[t + (size_t)NELEM] = x[t] - l;                 // high-pass
}

// ----------------------------------------------------------------------------

extern "C" void kernel_launch(void* const* d_in, const int* in_sizes, int n_in,
                              void* d_out, int out_size, void* d_ws, size_t ws_size,
                              hipStream_t stream) {
    const float* x  = (const float*)d_in[0];
    const float* w0 = (const float*)d_in[1];
    const float* b0 = (const float*)d_in[2];
    const float* w1 = (const float*)d_in[3];
    const float* b1 = (const float*)d_in[4];
    float* out = (float*)d_out;

    char* ws = (char*)d_ws;
    _Float16* x16 = (_Float16*)(ws + OFF_X16);
    _Float16* y0  = (_Float16*)(ws + OFF_Y0);
    float*    kern = (float*)(ws + OFF_KERN);
    _Float16* wp0 = (_Float16*)(ws + OFF_WP0);
    _Float16* wp1 = (_Float16*)(ws + OFF_WP1);

    k_x_to_nhwc_f16<<<NELEM / 256, 256, 0, stream>>>(x, x16);
    k_pack_w0<<<(4 * NCHUNKS * 32 + 255) / 256, 256, 0, stream>>>(w0, wp0);
    k_pack_w1<<<(NCHUNKS * 32 + 255) / 256, 256, 0, stream>>>(w1, wp1);
    k_conv0<<<NN * HH, 256, 0, stream>>>(x16, wp0, b0, y0);   // 1024 blocks, 8 waves each
    k_conv1<<<NN * HH, 256, 0, stream>>>(y0, wp1, b1, kern);
    k_softmax9<<<NPIX / 256, 256, 0, stream>>>(kern);
    k_filter<<<NELEM / 256, 256, 0, stream>>>(x, kern, out);
}
